// DSAttention_21045339750736
// MI455X (gfx1250) — matrix operand order repaired and checked by reading, not compiled, not run on hardware
//
#include <hip/hip_runtime.h>
#include <hip/hip_bf16.h>

typedef __attribute__((ext_vector_type(16))) _Float16 v16h;
typedef __attribute__((ext_vector_type(8)))  float    v8f;

// Problem constants (from reference)
#define CB 4
#define CL 2048
#define CH 8
#define CE 64
#define CD 64

#define WAVES 8
#define BLOCK (WAVES * 32)
#define QTILE (WAVES * 16)   // 128 query rows per workgroup

__global__ __launch_bounds__(BLOCK) void dsattn_fa_kernel(
    const float* __restrict__ Q, const float* __restrict__ K,
    const float* __restrict__ V, const float* __restrict__ Tau,
    const float* __restrict__ Delta, float* __restrict__ Out)
{
    __shared__ _Float16 ldsK[32 * 64];           // [key][e]      (f16)
    __shared__ _Float16 ldsV[64 * 32];           // [d][key]      (f16, transposed)
    __shared__ _Float16 ldsP[WAVES][16 * 32];    // per-wave P tile [row][key]

    const int bh   = blockIdx.y;          // 0 .. B*H-1
    const int b    = bh / CH;
    const int h    = bh % CH;
    const int qblk = blockIdx.x;          // 0 .. L/QTILE-1
    const int tid  = threadIdx.x;
    const int wave = tid >> 5;
    const int lane = tid & 31;
    const int half = lane >> 4;           // 0 or 1 (lane group)
    const int ln   = lane & 15;

    const int q0   = qblk * QTILE + wave * 16;   // wave's first query row
    const int qrow = q0 + ln;                    // A-layout row for this lane

    const float tau_b = Tau[b];
    const float scale = 0.125f;                  // 1/sqrt(64)

    const long rstride = (long)CH * CE;          // 512 floats per (b,l) row
    const float* Qb = Q + (long)b * CL * rstride + (long)h * CE;
    const float* Kb = K + (long)b * CL * rstride + (long)h * CE;
    const float* Vb = V + (long)b * CL * rstride + (long)h * CD;
    const float* db = Delta + (long)b * CL;
    float*       Ob = Out + (long)b * CL * rstride + (long)h * CD;

    // ---- Load Q (this wave's 16 rows) into 16-bit A-operand layout ----
    // lane<16 : K = {ebase..ebase+7, 16+ebase..16+ebase+7}, ebase = half*8
    v16h qa0, qa1;
    {
        const float* qr = Qb + (long)qrow * rstride;
        const int eb = half * 8;
        #pragma unroll
        for (int j = 0; j < 8; ++j) {
            qa0[j]     = (_Float16)qr[eb + j];
            qa0[8 + j] = (_Float16)qr[16 + eb + j];
            qa1[j]     = (_Float16)qr[32 + eb + j];
            qa1[8 + j] = (_Float16)qr[48 + eb + j];
        }
    }

    v8f acc[4] = {v8f{}, v8f{}, v8f{}, v8f{}};   // 16 rows x 64 cols (4 N-tiles)
    float m[8], l[8];
    #pragma unroll
    for (int r = 0; r < 8; ++r) { m[r] = -1.0e30f; l[r] = 0.0f; }

    const int nchunks = (qblk + 1) * (QTILE / 32);   // keys [0, (qblk+1)*QTILE)

    for (int c = 0; c < nchunks; ++c) {
        const int kbase = c * 32;
        __syncthreads();   // protect LDS tiles from previous iteration's readers

        // ---- cooperative K chunk load: global f32 -> ldsK [key][e] f16 ----
        for (int i = tid; i < (32 * 64) / 2; i += BLOCK) {
            const int idx = i * 2;
            const int key = idx >> 6, e = idx & 63;
            const float2 f = *(const float2*)(Kb + (long)(kbase + key) * rstride + e);
            ldsK[idx]     = (_Float16)f.x;
            ldsK[idx + 1] = (_Float16)f.y;
        }
        // ---- cooperative V chunk load: transpose into ldsV [d][key] f16 ----
        for (int i = tid; i < (32 * 64) / 2; i += BLOCK) {
            const int idx = i * 2;
            const int key = idx >> 6, d = idx & 63;
            const float2 f = *(const float2*)(Vb + (long)(kbase + key) * rstride + d);
            ldsV[d * 32 + key]       = (_Float16)f.x;
            ldsV[(d + 1) * 32 + key] = (_Float16)f.y;
        }
        __syncthreads();

        // ---- QK^T: two 16x16 score tiles (keys kbase+0..15, kbase+16..31) ----
        v8f s[2];
        #pragma unroll
        for (int t = 0; t < 2; ++t) {
            // B operand: lane column = key (t*16+ln); K-dim e = {half*8..+7, 16+half*8..+7} (table order)
            const _Float16* kp = &ldsK[(t * 16 + ln) * 64 + half * 8];
            v16h kb0, kb1;
            #pragma unroll
            for (int j = 0; j < 16; ++j) {
                kb0[j] = kp[(j & 7) + ((j >> 3) << 4)];        // e = half*8 + table offset
                kb1[j] = kp[32 + (j & 7) + ((j >> 3) << 4)];   // e = 32 + half*8 + table offset
            }
            v8f cz = {};
            cz = __builtin_amdgcn_wmma_f32_16x16x32_f16(false, qa0, false, kb0,
                                                        (short)0, cz, false, false);
            cz = __builtin_amdgcn_wmma_f32_16x16x32_f16(false, qa1, false, kb1,
                                                        (short)0, cz, false, false);
            const int col = kbase + t * 16 + ln;
            const float dlt = db[col];
            #pragma unroll
            for (int r = 0; r < 8; ++r) {
                const int row = q0 + r + 8 * half;        // C layout: M = r + 8*half
                const float v = (cz[r] * tau_b + dlt) * scale;
                s[t][r] = (col <= row) ? v : -1.0e30f;    // causal mask
            }
        }

        // ---- online softmax (row stats reduced across 16-lane column groups) ----
        #pragma unroll
        for (int r = 0; r < 8; ++r) {
            float mx = fmaxf(s[0][r], s[1][r]);
            #pragma unroll
            for (int o = 8; o >= 1; o >>= 1)
                mx = fmaxf(mx, __shfl_xor(mx, o, 32));
            const float mnew = fmaxf(m[r], mx);
            const float corr = __expf(m[r] - mnew);
            m[r] = mnew;
            const float p0 = __expf(s[0][r] - mnew);
            const float p1 = __expf(s[1][r] - mnew);
            float rs = p0 + p1;
            #pragma unroll
            for (int o = 8; o >= 1; o >>= 1)
                rs += __shfl_xor(rs, o, 32);
            l[r] = l[r] * corr + rs;
            #pragma unroll
            for (int nt = 0; nt < 4; ++nt)
                acc[nt][r] *= corr;
            s[0][r] = p0;
            s[1][r] = p1;
        }

        // ---- re-layout P: C layout -> 16-bit A layout via per-wave LDS tile ----
        _Float16* pb = ldsP[wave];
        #pragma unroll
        for (int t = 0; t < 2; ++t)
            #pragma unroll
            for (int r = 0; r < 8; ++r)
                pb[(r + 8 * half) * 32 + t * 16 + ln] = (_Float16)s[t][r];

        asm volatile("s_wait_dscnt 0" ::: "memory");   // wave-local store->load ordering

        v16h pa;
        {
            const _Float16* pr = &pb[ln * 32 + half * 8];
            #pragma unroll
            for (int j = 0; j < 8; ++j) { pa[j] = pr[j]; pa[8 + j] = pr[16 + j]; }
        }

        // ---- P * V : 4 N-tiles of 16 output columns, K = 32 keys ----
        #pragma unroll
        for (int nt = 0; nt < 4; ++nt) {
            const _Float16* vp = &ldsV[(nt * 16 + ln) * 32 + half * 8];
            v16h vb;
            #pragma unroll
            for (int j = 0; j < 16; ++j) vb[j] = vp[(j & 7) + ((j >> 3) << 4)];
            acc[nt] = __builtin_amdgcn_wmma_f32_16x16x32_f16(false, pa, false, vb,
                                                             (short)0, acc[nt],
                                                             false, false);
        }
    }

    // ---- epilogue: divide by row sums, store f32 ----
    #pragma unroll
    for (int r = 0; r < 8; ++r) {
        const float inv = 1.0f / l[r];
        const int row = q0 + r + 8 * half;
        float* op = Ob + (long)row * rstride;
        #pragma unroll
        for (int nt = 0; nt < 4; ++nt)
            op[nt * 16 + ln] = acc[nt][r] * inv;
    }
}

extern "C" void kernel_launch(void* const* d_in, const int* in_sizes, int n_in,
                              void* d_out, int out_size, void* d_ws, size_t ws_size,
                              hipStream_t stream) {
    (void)in_sizes; (void)n_in; (void)out_size; (void)d_ws; (void)ws_size;
    const float* Q     = (const float*)d_in[0];
    const float* K     = (const float*)d_in[1];
    const float* V     = (const float*)d_in[2];
    const float* tau   = (const float*)d_in[3];
    const float* delta = (const float*)d_in[4];
    float* Out = (float*)d_out;

    dim3 grid(CL / QTILE, CB * CH);   // (16, 32)
    dim3 block(BLOCK);                // 256 threads = 8 waves
    hipLaunchKernelGGL(dsattn_fa_kernel, grid, block, 0, stream,
                       Q, K, V, tau, delta, Out);
}
